// Fp8Attention_77266461655619
// MI455X (gfx1250) — 16-bit and 8-bit matrix operand order repaired and checked by reading, compiled, not run on hardware
//
#include <hip/hip_runtime.h>
#include <hip/hip_bf16.h>

typedef __attribute__((ext_vector_type(16))) int      v16i;
typedef __attribute__((ext_vector_type(4)))  int      v4i;
typedef __attribute__((ext_vector_type(8)))  float    v8f;
typedef __attribute__((ext_vector_type(16))) _Float16 v16h;
typedef __attribute__((ext_vector_type(2)))  _Float16 v2h;

#define SOFTMAX_SCALE 0.08838834764831845f
#define SEQ 1024
#define NHEAD 32
#define HDIM 128
#define NBATCH 4
#define NELEM (NBATCH * SEQ * NHEAD * HDIM)

#if __has_builtin(__builtin_amdgcn_global_load_async_to_lds_b128)
#define HAVE_ASYNC_LDS 1
#endif
#if __has_builtin(__builtin_amdgcn_cvt_pk_f16_fp8)
#define HAVE_CVT_F16_FP8 1
#endif


__device__ inline unsigned f32_to_e4m3_sw(float f) {
    unsigned u = __float_as_uint(f);
    unsigned s = (u >> 31) << 7;
    float a = fabsf(f);
    if (a >= 448.f) return s | 0x7Eu;
    int ex = (int)((u >> 23) & 0xFF) - 127;
    if (ex < -6) {
        unsigned q = (unsigned)rintf(a * 512.f);
        return s | q;
    }
    unsigned mant = u & 0x7FFFFF;
    unsigned keep = mant >> 20;
    unsigned rest = mant & 0xFFFFF;
    keep += (rest > 0x80000u) || (rest == 0x80000u && (keep & 1u));
    unsigned code = ((unsigned)(ex + 7) << 3) + keep;
    if (code > 0x7Eu) code = 0x7Eu;
    return s | code;
}

__device__ inline unsigned pk4_e4m3(float x, float y, float z, float w) {
#if __has_builtin(__builtin_amdgcn_cvt_pk_fp8_f32)
    int lo = __builtin_amdgcn_cvt_pk_fp8_f32(x, y, 0, false);
    int r  = __builtin_amdgcn_cvt_pk_fp8_f32(z, w, lo, true);
    return (unsigned)r;
#else
    return f32_to_e4m3_sw(x) | (f32_to_e4m3_sw(y) << 8) |
           (f32_to_e4m3_sw(z) << 16) | (f32_to_e4m3_sw(w) << 24);
#endif
}

__device__ inline unsigned short fp8_to_h_bits(unsigned c) {
    unsigned s = (c & 0x80u) << 8;
    unsigned e = (c >> 3) & 15u, m = c & 7u;
    if (e) return (unsigned short)(s | ((e + 8u) << 10) | (m << 7));
    if (!m) return (unsigned short)s;
    int k = 31 - __clz((int)m);
    return (unsigned short)(s | ((unsigned)(k + 6) << 10) | ((m ^ (1u << k)) << (10 - k)));
}

__device__ inline void cvt_word_to_h4(unsigned w, unsigned short* o) {
#ifdef HAVE_CVT_F16_FP8
    v2h lo = __builtin_amdgcn_cvt_pk_f16_fp8((short)(w & 0xFFFFu));
    v2h hi = __builtin_amdgcn_cvt_pk_f16_fp8((short)(w >> 16));
    __builtin_memcpy(o + 0, &lo, 4);
    __builtin_memcpy(o + 2, &hi, 4);
#else
    o[0] = fp8_to_h_bits(w & 255u);
    o[1] = fp8_to_h_bits((w >> 8) & 255u);
    o[2] = fp8_to_h_bits((w >> 16) & 255u);
    o[3] = fp8_to_h_bits(w >> 24);
#endif
}

__device__ __forceinline__ void wait_async0() {
#if __has_builtin(__builtin_amdgcn_s_wait_asynccnt)
    __builtin_amdgcn_s_wait_asynccnt(0);
#else
    asm volatile("s_wait_asynccnt 0x0" ::: "memory");
#endif
}


__global__ __launch_bounds__(256) void absmax_kernel(
    const float* __restrict__ x0, const float* __restrict__ x1,
    const float* __restrict__ x2, unsigned* __restrict__ amax, int n) {
    const float* x = (blockIdx.y == 0) ? x0 : (blockIdx.y == 1) ? x1 : x2;
    const float4* xv = (const float4*)x;
    int nv = n >> 2;
    float m = 0.f;
    for (int i = blockIdx.x * blockDim.x + threadIdx.x; i < nv;
         i += gridDim.x * blockDim.x) {
        float4 t = xv[i];
        m = fmaxf(m, fmaxf(fmaxf(fabsf(t.x), fabsf(t.y)),
                           fmaxf(fabsf(t.z), fabsf(t.w))));
    }
    for (int d = 16; d; d >>= 1) m = fmaxf(m, __shfl_xor(m, d));
    __shared__ float red[8];
    int lane = threadIdx.x & 31, w = threadIdx.x >> 5;
    if (!lane) red[w] = m;
    __syncthreads();
    if (threadIdx.x < 8) {
        m = red[threadIdx.x];
        for (int d = 4; d; d >>= 1) m = fmaxf(m, __shfl_xor(m, d));
        if (!threadIdx.x) atomicMax(amax + blockIdx.y, __float_as_uint(m));
    }
}


__global__ __launch_bounds__(256) void quant_kernel(
    const float* __restrict__ x0, const float* __restrict__ x1,
    const float* __restrict__ x2, unsigned char* __restrict__ o0,
    unsigned char* __restrict__ o1, unsigned char* __restrict__ o2,
    const unsigned* __restrict__ amax, int n) {
    int t = blockIdx.y;
    const float* x = (t == 0) ? x0 : (t == 1) ? x1 : x2;
    unsigned char* o = (t == 0) ? o0 : (t == 1) ? o1 : o2;
    float inv = 448.f / __uint_as_float(amax[t]);
    int nv = n >> 4;
    for (int i = blockIdx.x * blockDim.x + threadIdx.x; i < nv;
         i += gridDim.x * blockDim.x) {
        const float4* xv = (const float4*)x + (size_t)i * 4;
        uint4 r;
        float4 a = xv[0], b = xv[1], c = xv[2], d = xv[3];
        r.x = pk4_e4m3(a.x * inv, a.y * inv, a.z * inv, a.w * inv);
        r.y = pk4_e4m3(b.x * inv, b.y * inv, b.z * inv, b.w * inv);
        r.z = pk4_e4m3(c.x * inv, c.y * inv, c.z * inv, c.w * inv);
        r.w = pk4_e4m3(d.x * inv, d.y * inv, d.z * inv, d.w * inv);
        ((uint4*)o)[i] = r;
    }
}


#ifdef HAVE_ASYNC_LDS
__device__ __forceinline__ void v_issue_async(
    const unsigned char* __restrict__ v8, uint4* dst, int b, int h, int ch,
    int tid) {
    int tok = tid >> 3, seg = tid & 7;
#pragma unroll
    for (int g = 0; g < 2; ++g) {
        const unsigned char* gsrc =
            v8 + ((size_t)(b * SEQ + ch * 32 + tok + g * 16) * NHEAD + h) * HDIM +
            seg * 16;
        __builtin_amdgcn_global_load_async_to_lds_b128(
            (__attribute__((address_space(1))) v4i*)gsrc,
            (__attribute__((address_space(3))) v4i*)(dst + tok * 8 + seg + g * 128),
            0, 0);
    }
}
#endif

__global__ __launch_bounds__(128) void flash_fp8_kernel(
    const unsigned char* __restrict__ q8, const unsigned char* __restrict__ k8,
    const unsigned char* __restrict__ v8, const unsigned* __restrict__ amax,
    float* __restrict__ out) {
    __shared__ unsigned short vT[128 * 40];
    __shared__ unsigned short plds[4][16 * 36];
#ifdef HAVE_ASYNC_LDS
    __shared__ uint4 vraw[2][256];
#endif

    const int blk = blockIdx.x;
    const int qt = blk & 15;
    const int h  = (blk >> 4) & 31;
    const int b  = blk >> 9;
    const int qb = qt * 64;

    const int tid  = threadIdx.x;
    const int wave = tid >> 5;
    const int lane = tid & 31;
    const int l16  = lane & 15;
    const int hi   = lane >> 4;

    const float qs = __uint_as_float(amax[0]) * (1.f / 448.f);
    const float ks = __uint_as_float(amax[1]) * (1.f / 448.f);
    const float vs = __uint_as_float(amax[2]) * (1.f / 448.f);
    const float sm = SOFTMAX_SCALE * qs * ks;

    const int aRow = qb + wave * 16 + l16;
    const unsigned char* qbase = q8 + ((size_t)(b * SEQ + aRow) * NHEAD + h) * HDIM;
    v16i aq;
#pragma unroll
    for (int k = 0; k < 8; ++k) {
        int off = (k >> 1) * 32 + hi * 16 + (k & 1) * 8;
        int2 t = *(const int2*)(qbase + off);
        aq[2 * k] = t.x;
        aq[2 * k + 1] = t.y;
    }

    v8f acc[8];
    v8f cz = {};
#pragma unroll
    for (int i = 0; i < 8; ++i) acc[i] = cz;
    float mrow[8], lrow[8];
#pragma unroll
    for (int i = 0; i < 8; ++i) { mrow[i] = -3.0e38f; lrow[i] = 0.f; }

    const int rowD_base = qb + wave * 16 + hi * 8;

    const int nch = (qb >> 5) + 2;

#ifdef HAVE_ASYNC_LDS
    v_issue_async(v8, &vraw[0][0], b, h, 0, tid);
#endif

    for (int ch = 0; ch < nch; ++ch) {
#ifdef HAVE_ASYNC_LDS
        wait_async0();
        __syncthreads();
        if (ch + 1 < nch)
            v_issue_async(v8, &vraw[(ch + 1) & 1][0], b, h, ch + 1, tid);
        {
            int vtok  = tid >> 2;
            int dbase = (tid & 3) * 32;
            const uint4* src = &vraw[ch & 1][vtok * 8 + (tid & 3) * 2];
#pragma unroll
            for (int g = 0; g < 2; ++g) {
                uint4 wv = src[g];
                unsigned words[4] = {wv.x, wv.y, wv.z, wv.w};
#pragma unroll
                for (int j = 0; j < 4; ++j) {
                    unsigned short h4[4];
                    cvt_word_to_h4(words[j], h4);
#pragma unroll
                    for (int bi = 0; bi < 4; ++bi)
                        vT[(dbase + g * 16 + j * 4 + bi) * 40 + vtok] = h4[bi];
                }
            }
        }
#else
        __syncthreads();
        {
            int vtok  = tid >> 2;
            int dbase = (tid & 3) * 32;
            const unsigned char* vb =
                v8 + ((size_t)(b * SEQ + ch * 32 + vtok) * NHEAD + h) * HDIM + dbase;
#pragma unroll
            for (int g = 0; g < 2; ++g) {
                uint4 wv = *(const uint4*)(vb + g * 16);
                unsigned words[4] = {wv.x, wv.y, wv.z, wv.w};
#pragma unroll
                for (int j = 0; j < 4; ++j) {
                    unsigned short h4[4];
                    cvt_word_to_h4(words[j], h4);
#pragma unroll
                    for (int bi = 0; bi < 4; ++bi)
                        vT[(dbase + g * 16 + j * 4 + bi) * 40 + vtok] = h4[bi];
                }
            }
        }
#endif
        __syncthreads();

        const unsigned char* kc0 =
            k8 + ((size_t)(b * SEQ + ch * 32 + l16) * NHEAD + h) * HDIM;
        const unsigned char* kc1 = kc0 + (size_t)16 * NHEAD * HDIM;
        v16i bk0, bk1;
#pragma unroll
        for (int j = 0; j < 4; ++j) {
            uint4 t0 = *(const uint4*)(kc0 + j * 32 + hi * 16);
            uint4 t1 = *(const uint4*)(kc1 + j * 32 + hi * 16);
            bk0[4 * j + 0] = (int)t0.x; bk0[4 * j + 1] = (int)t0.y;
            bk0[4 * j + 2] = (int)t0.z; bk0[4 * j + 3] = (int)t0.w;
            bk1[4 * j + 0] = (int)t1.x; bk1[4 * j + 1] = (int)t1.y;
            bk1[4 * j + 2] = (int)t1.z; bk1[4 * j + 3] = (int)t1.w;
        }
        if (ch + 1 < nch)
            __builtin_prefetch(kc0 + (size_t)32 * NHEAD * HDIM, 0, 3);

        v8f s0 = __builtin_amdgcn_wmma_f32_16x16x128_fp8_fp8(aq, bk0, (short)0, cz, false, false);
        v8f s1 = __builtin_amdgcn_wmma_f32_16x16x128_fp8_fp8(aq, bk1, (short)0, cz, false, false);

        float sv0[8], sv1[8], rmax[8];
        const int c0 = ch * 32 + l16;
        const int c1 = c0 + 16;
#pragma unroll
        for (int i = 0; i < 8; ++i) {
            int rq = rowD_base + i;
            float a0 = (c0 <= rq) ? s0[i] * sm : -3.0e38f;
            float a1 = (c1 <= rq) ? s1[i] * sm : -3.0e38f;
            sv0[i] = a0; sv1[i] = a1;
            float r = fmaxf(a0, a1);
            r = fmaxf(r, __shfl_xor(r, 1));
            r = fmaxf(r, __shfl_xor(r, 2));
            r = fmaxf(r, __shfl_xor(r, 4));
            r = fmaxf(r, __shfl_xor(r, 8));
            rmax[i] = r;
        }

        float alpha[8];
#pragma unroll
        for (int i = 0; i < 8; ++i) {
            float mnew = fmaxf(mrow[i], rmax[i]);
            alpha[i] = __expf(mrow[i] - mnew);
            mrow[i] = mnew;
            float p0 = __expf(sv0[i] - mnew);
            float p1 = __expf(sv1[i] - mnew);
            float rs = p0 + p1;
            rs += __shfl_xor(rs, 1);
            rs += __shfl_xor(rs, 2);
            rs += __shfl_xor(rs, 4);
            rs += __shfl_xor(rs, 8);
            lrow[i] = lrow[i] * alpha[i] + rs;
            int prow = hi * 8 + i;
            _Float16 h0 = (_Float16)p0, h1 = (_Float16)p1;
            unsigned short u0, u1;
            __builtin_memcpy(&u0, &h0, 2);
            __builtin_memcpy(&u1, &h1, 2);
            plds[wave][prow * 36 + l16] = u0;
            plds[wave][prow * 36 + 16 + l16] = u1;
        }
#pragma unroll
        for (int dt = 0; dt < 8; ++dt)
#pragma unroll
            for (int i = 0; i < 8; ++i) acc[dt][i] = acc[dt][i] * alpha[i];

        v16h ap;
        {
            const unsigned short* pr = &plds[wave][l16 * 36];
#pragma unroll
            for (int j = 0; j < 8; ++j) {
                int tb = (j >> 2) * 16 + (j & 3) * 2 + hi * 8;
                unsigned short u0 = pr[tb], u1 = pr[tb + 1];
                _Float16 h0, h1;
                __builtin_memcpy(&h0, &u0, 2);
                __builtin_memcpy(&h1, &u1, 2);
                ap[2 * j] = h0;
                ap[2 * j + 1] = h1;
            }
        }

#pragma unroll
        for (int dt = 0; dt < 8; ++dt) {
            v16h bv;
            const unsigned short* vr = &vT[(dt * 16 + l16) * 40];
#pragma unroll
            for (int t2 = 0; t2 < 16; ++t2) {
                unsigned short u = vr[(t2 >> 3) * 16 + hi * 8 + (t2 & 7)];
                _Float16 hh;
                __builtin_memcpy(&hh, &u, 2);
                bv[t2] = hh;
            }
            acc[dt] = __builtin_amdgcn_wmma_f32_16x16x32_f16(
                false, ap, false, bv, (short)0, acc[dt], false, false);
        }
    }

#pragma unroll
    for (int i = 0; i < 8; ++i) {
        float invl = vs / lrow[i];
        int row = rowD_base + i;
        float* ob = out + ((size_t)(b * SEQ + row) * NHEAD + h) * HDIM + l16;
#pragma unroll
        for (int dt = 0; dt < 8; ++dt) ob[dt * 16] = acc[dt][i] * invl;
    }
}


extern "C" void kernel_launch(void* const* d_in, const int* in_sizes, int n_in,
                              void* d_out, int out_size, void* d_ws, size_t ws_size,
                              hipStream_t stream) {
    const float* q = (const float*)d_in[0];
    const float* k = (const float*)d_in[1];
    const float* v = (const float*)d_in[2];
    float* out = (float*)d_out;

    unsigned* amax = (unsigned*)d_ws;
    unsigned char* base = (unsigned char*)d_ws + 256;
    unsigned char* q8 = base;
    unsigned char* k8 = q8 + (size_t)NELEM;
    unsigned char* v8 = k8 + (size_t)NELEM;

    (void)hipMemsetAsync(d_ws, 0, 16, stream);

    dim3 g1(1024, 3);
    absmax_kernel<<<g1, 256, 0, stream>>>(q, k, v, amax, NELEM);

    dim3 g2(4096, 3);
    quant_kernel<<<g2, 256, 0, stream>>>(q, k, v, q8, k8, v8, amax, NELEM);

    flash_fp8_kernel<<<NBATCH * NHEAD * (SEQ / 64), 128, 0, stream>>>(
        q8, k8, v8, amax, out);
}
